// CUTS_Plus_Net_4355096838439
// MI455X (gfx1250) — compile-verified, matrix operand order checked by reading, not run on hardware
//
#include <hip/hip_runtime.h>
#include <math.h>

// ---- problem constants (from reference) ----
#define BB   8
#define NN   1024
#define SS   64
#define HCH  32
#define NT   32        // n-columns per workgroup
#define THREADS 256    // 8 wave32 waves

typedef __attribute__((ext_vector_type(16))) __bf16 v16bf;
typedef __attribute__((ext_vector_type(2)))  __bf16 bf16x2;
typedef __attribute__((ext_vector_type(8)))  float  v8f;

typedef __attribute__((address_space(1))) int gas_int;   // global
typedef __attribute__((address_space(3))) int las_int;   // LDS

#if __has_builtin(__builtin_amdgcn_global_load_async_to_lds_b32)
#define HAVE_ASYNC_LDS 1
#else
#define HAVE_ASYNC_LDS 0
#endif

__device__ __forceinline__ float lrelu(float a) { return a >= 0.f ? a : 0.01f * a; }
__device__ __forceinline__ float fast_rcp(float a) { return __builtin_amdgcn_rcpf(a); }

// pack two f32 -> packed bf16x2 via native converts (v_cvt_pk_bf16_f32 path)
__device__ __forceinline__ void set_pair(v16bf& v, int j, float lo, float hi) {
    v[2 * j]     = (__bf16)lo;
    v[2 * j + 1] = (__bf16)hi;
}
__device__ __forceinline__ unsigned pack_pair(float lo, float hi) {
    bf16x2 p; p[0] = (__bf16)lo; p[1] = (__bf16)hi;
    return __builtin_bit_cast(unsigned, p);
}

__global__ __launch_bounds__(THREADS)
void cuts_gru_wmma(const float* __restrict__ x,    // (B,N,S)
                   const float* __restrict__ g,    // (B,N,N)  g[b,n,m]
                   const float* __restrict__ Wu,   // (32,1024)
                   const float* __restrict__ bu,
                   const float* __restrict__ Wc,
                   const float* __restrict__ bc,
                   const float* __restrict__ h0,   // (32,1024)
                   const float* __restrict__ W1, const float* __restrict__ b1,
                   const float* __restrict__ W2, const float* __restrict__ b2,
                   const float* __restrict__ Wd, const float* __restrict__ bd,
                   float* __restrict__ out)        // (B,N)
{
    __shared__ unsigned sW[64 * 512];      // 128KB: [row 0..63][k-pair] packed bf16x2
    __shared__ float sx[NN];               // 4KB: x[b,:,s]
    __shared__ float sStage[64 * NT];      // 8KB: u rows 0-31 / c rows 32-63
    __shared__ float sh[HCH * NT];         // 4KB: hidden state tile
    __shared__ float sbias[64];            // bu ++ bc

    const int tid = threadIdx.x;
    const int b  = blockIdx.y;
    const int n0 = blockIdx.x * NT;

    // ---- one-time: W (both gates) -> bf16 LDS; biases; h init ----
    for (int idx = tid; idx < 64 * 512; idx += THREADS) {
        const int row = idx >> 9, p = idx & 511;
        const float* src = (row < HCH) ? (Wu + row * NN) : (Wc + (row - HCH) * NN);
        const float2 w = *(const float2*)(src + 2 * p);
        sW[idx] = pack_pair(w.x, w.y);     // single v_cvt_pk_bf16_f32
    }
    if (tid < 64) sbias[tid] = (tid < HCH) ? bu[tid] : bc[tid - HCH];
    for (int idx = tid; idx < HCH * NT; idx += THREADS)
        sh[idx] = h0[(idx / NT) * NN + n0 + (idx % NT)];
    __syncthreads();

    // ---- wave / lane -> tile mapping (CDNA5 wave32 WMMA layouts) ----
    const int lane  = tid & 31;
    const int wid   = tid >> 5;
    const int row0  = (wid >> 1) * 16;     // 4 row-tiles: waves 0-3 u-gate, 4-7 c-gate
    const int col0  = (wid & 1) * 16;      // 2 col-tiles
    const int mrow  = lane & 15;
    const int khalf = (lane >> 4) * 8;     // 16-bit A/B 16x32 layout: hi lanes hold K+8
    const int aRow  = row0 + mrow;
    const unsigned* sWrow = sW + aRow * 512;
    const float* grow = g + ((size_t)(b * NN + n0 + col0 + mrow)) * NN;

    for (int s = 0; s < SS; ++s) {
        // ---- fetch x[b,:,s] into LDS (CDNA5 async-to-LDS path) ----
#if HAVE_ASYNC_LDS
        for (int idx = tid; idx < NN; idx += THREADS) {
            __builtin_amdgcn_global_load_async_to_lds_b32(
                (gas_int*)(float*)(x + (size_t)(b * NN + idx) * SS + s),
                (las_int*)(sx + idx), 0, 0);
        }
#if __has_builtin(__builtin_amdgcn_s_wait_asynccnt)
        __builtin_amdgcn_s_wait_asynccnt(0);
#else
        asm volatile("s_wait_asynccnt 0" ::: "memory");
#endif
#else
        for (int idx = tid; idx < NN; idx += THREADS)
            sx[idx] = x[(size_t)(b * NN + idx) * SS + s];
#endif
        __syncthreads();

        v8f acc = {};
        #pragma unroll 2
        for (int k0 = 0; k0 < NN; k0 += 32) {
            const int kb = k0 + khalf;                    // float index base
            // A fragment: 2x ds_load_b128 (16B-aligned: kb/2 multiple of 4)
            union { uint4 q[2]; v16bf v; } A;
            A.q[0] = *(const uint4*)(sWrow + (kb >> 1));
            A.q[1] = *(const uint4*)(sWrow + (kb >> 1) + 8);
            // B fragment: g row (global_load_b128 x4, L2-hot) scaled by x (ds b128 x4)
            const float* gp = grow + kb;
            const float4 g0 = *(const float4*)(gp + 0);
            const float4 g1 = *(const float4*)(gp + 4);
            const float4 g2 = *(const float4*)(gp + 16);
            const float4 g3 = *(const float4*)(gp + 20);
            const float4 x0 = *(const float4*)(sx + kb + 0);
            const float4 x1 = *(const float4*)(sx + kb + 4);
            const float4 x2 = *(const float4*)(sx + kb + 16);
            const float4 x3 = *(const float4*)(sx + kb + 20);
            v16bf Bv;
            set_pair(Bv, 0, g0.x * x0.x, g0.y * x0.y);
            set_pair(Bv, 1, g0.z * x0.z, g0.w * x0.w);
            set_pair(Bv, 2, g1.x * x1.x, g1.y * x1.y);
            set_pair(Bv, 3, g1.z * x1.z, g1.w * x1.w);
            set_pair(Bv, 4, g2.x * x2.x, g2.y * x2.y);
            set_pair(Bv, 5, g2.z * x2.z, g2.w * x2.w);
            set_pair(Bv, 6, g3.x * x3.x, g3.y * x3.y);
            set_pair(Bv, 7, g3.z * x3.z, g3.w * x3.w);
            acc = __builtin_amdgcn_wmma_f32_16x16x32_bf16(
                false, A.v, false, Bv, (short)0, acc, false, false);
        }

        // ---- bias + activation (wave-uniform gate select), stage to LDS ----
        // C/D layout: VGPR v -> M = v (+8 for hi lanes), N = lane&15
        const int col = col0 + (lane & 15);
        const int rbase = row0 + 8 * (lane >> 4);
        if (row0 < HCH) {       // u-gate: sigmoid = rcp(1+e^{-v})
            #pragma unroll
            for (int v = 0; v < 8; ++v) {
                const int row = rbase + v;
                const float val = acc[v] + sbias[row];
                sStage[row * NT + col] = fast_rcp(1.f + __expf(-val));
            }
        } else {                // c-gate: tanh = 1 - 2*rcp(1+e^{2v}) (branch-free, saturating)
            #pragma unroll
            for (int v = 0; v < 8; ++v) {
                const int row = rbase + v;
                const float val = acc[v] + sbias[row];
                sStage[row * NT + col] = 1.f - 2.f * fast_rcp(1.f + __expf(2.f * val));
            }
        }
        __syncthreads();

        // ---- GRU scan: h = u*h + (1-u)*c ----
        for (int idx = tid; idx < HCH * NT; idx += THREADS) {
            const float uu = sStage[idx];
            const float cc = sStage[HCH * NT + idx];
            sh[idx] = uu * sh[idx] + (1.f - uu) * cc;
        }
        __syncthreads();
    }

    // ---- prediction head: one column per thread (tiny; L2-resident weights) ----
    if (tid < NT) {
        const int nl = tid, n = n0 + nl;
        float hcol[HCH], x1[HCH], x2[HCH];
        #pragma unroll
        for (int i = 0; i < HCH; ++i) hcol[i] = sh[i * NT + nl];
        for (int i = 0; i < HCH; ++i) {
            float a = b1[i];
            for (int j = 0; j < HCH; ++j) a += W1[i * HCH + j] * hcol[j];
            x1[i] = lrelu(a);
        }
        for (int i = 0; i < HCH; ++i) {
            float a = b2[i];
            for (int j = 0; j < HCH; ++j) a += W2[i * 64 + j] * x1[j];
            for (int j = 0; j < HCH; ++j) a += W2[i * 64 + HCH + j] * hcol[j];
            x2[i] = lrelu(a);
        }
        float a = bd[n];
        for (int j = 0; j < HCH; ++j) a += Wd[n * 64 + j] * x2[j];
        for (int j = 0; j < HCH; ++j) a += Wd[n * 64 + HCH + j] * hcol[j];
        out[(size_t)b * NN + n] = a;   // (b,n,1,1) flat
    }
}

extern "C" void kernel_launch(void* const* d_in, const int* in_sizes, int n_in,
                              void* d_out, int out_size, void* d_ws, size_t ws_size,
                              hipStream_t stream) {
    (void)in_sizes; (void)n_in; (void)d_ws; (void)ws_size; (void)out_size;
    const float* x  = (const float*)d_in[0];
    // d_in[1] = mask (unused); d_in[3..4] = Wf,bf (mathematically dead)
    const float* g  = (const float*)d_in[2];
    const float* Wu = (const float*)d_in[5];
    const float* bu = (const float*)d_in[6];
    const float* Wc = (const float*)d_in[7];
    const float* bc = (const float*)d_in[8];
    const float* h0 = (const float*)d_in[9];
    const float* W1 = (const float*)d_in[10];
    const float* b1 = (const float*)d_in[11];
    const float* W2 = (const float*)d_in[12];
    const float* b2 = (const float*)d_in[13];
    const float* Wd = (const float*)d_in[14];
    const float* bd = (const float*)d_in[15];
    float* out = (float*)d_out;

    dim3 grid(NN / NT, BB);   // 32 x 8 = 256 workgroups
    cuts_gru_wmma<<<grid, dim3(THREADS), 0, stream>>>(
        x, g, Wu, bu, Wc, bc, h0, W1, b1, W2, b2, Wd, bd, out);
}
